// AliAttention_44375602102815
// MI455X (gfx1250) — matrix operand order repaired and checked by reading, not compiled, not run on hardware
//
#include <hip/hip_runtime.h>
#include <hip/hip_bf16.h>
#include <math.h>

// ---------------------------------------------------------------------------
// Types for CDNA5 WMMA (wave32)
// ---------------------------------------------------------------------------
typedef __bf16 bf16;
typedef __bf16 bf16x8  __attribute__((ext_vector_type(8)));
typedef __bf16 bf16x16 __attribute__((ext_vector_type(16)));
typedef float  f32x8   __attribute__((ext_vector_type(8)));
typedef unsigned int u32x4 __attribute__((ext_vector_type(4)));
typedef int i32x4 __attribute__((ext_vector_type(4)));
typedef int i32x8 __attribute__((ext_vector_type(8)));

#define GM 4096
#define GN 512
#define GK 512
#define SEQ 2048
#define DKH 64

__device__ __forceinline__ f32x8 wmma_bf16(bf16x16 a, bf16x16 b, f32x8 c) {
  return __builtin_amdgcn_wmma_f32_16x16x32_bf16(
      false, a, false, b, (short)0, c, false, false);
}

__device__ __forceinline__ f32x8 zero8() {
  f32x8 z;
#pragma unroll
  for (int i = 0; i < 8; ++i) z[i] = 0.0f;
  return z;
}

// Fragment from two 16-byte chunks (works for LDS or global bf16).
// A-layout (16x32): lane holds its row's K {16h..16h+15} contiguously.
// B-layout (32x16): lane holds its column's K {16h..16h+15} contiguously.
__device__ __forceinline__ bf16x16 ld_frag2(const bf16* p0, const bf16* p1) {
  bf16x8 a = *(const bf16x8*)p0;
  bf16x8 b = *(const bf16x8*)p1;
  bf16x16 r;
#pragma unroll
  for (int i = 0; i < 8; ++i) { r[i] = a[i]; r[i + 8] = b[i]; }
  return r;
}

// ---------------------------------------------------------------------------
// Tensor Data Mover: 2D tile (64 rows x 64 bf16) global -> LDS, padded so
// LDS pitch = 72 bf16 (pad 4 DWORDs after every 32 DWORDs = one 128B row).
// D# built per CDNA5 ISA. Issue from ONE wave.
// This toolchain exposes the 6-arg builtin (clang-23 / therock-10.0 form).
// ---------------------------------------------------------------------------
__device__ __forceinline__ void tdm_load_tile64(unsigned lds_off, const bf16* g) {
  const unsigned long long ga = (unsigned long long)(uintptr_t)g;
  u32x4 g0;
  g0[0] = 1u;                                   // count=1, user mode
  g0[1] = lds_off;                              // lds_addr (bytes)
  g0[2] = (unsigned)(ga & 0xffffffffu);         // global_addr[95:64]
  g0[3] = (unsigned)((ga >> 32) & 0x01ffffffu)  // global_addr[120:96]
        | (2u << 30);                           // type = 2 ("image")
  i32x8 g1;
  g1[0] = (1 << 16)      // data_size = 1 -> 2 bytes
        | (1 << 20)      // pad_enable
        | (4 << 22)      // pad_interval: code 4 = 32 DWORDs (one 128B row)
        | (3 << 25);     // pad_amount:   code 3 = 4 DWORDs (16B) -> pitch 144B
  g1[1] = (64 << 16);    // tensor_dim0 = 64 (bits 79:48, low part)
  g1[2] = (64 << 16);    // tensor_dim0 hi=0 | tensor_dim1 = 64 (bits 95:80)
  g1[3] = (64 << 16);    // tensor_dim1 hi=0 | tile_dim0 = 64 (bits 127:112)
  g1[4] = 64;            // tile_dim1 = 64, tile_dim2 = 0
  g1[5] = 64;            // tensor_dim0_stride = 64 (bits 191:160)
  g1[6] = 0;             // stride hi, tensor_dim1_stride lo
  g1[7] = 0;
  const i32x4 z4 = {0, 0, 0, 0};              // groups 2/3 unused (2D tensor)
  const i32x8 z8 = {0, 0, 0, 0, 0, 0, 0, 0};  // trailing group (zero-filled)
  __builtin_amdgcn_tensor_load_to_lds(g0, g1, z4, z4, z8, 0);
}

// ---------------------------------------------------------------------------
// GEMM: C(4096x512) = A(4096x512) @ W(512x512) + bias. W is f32; A/C types
// templated (f32 or bf16). Block = 128 thr (4 waves, 2x2); tile 128x128;
// each wave 64x64 = 16 WMMA per K-step of 32.
// ---------------------------------------------------------------------------
__device__ __forceinline__ void stage_a(bf16* As, const float* A,
                                        int m0, int k0, int t) {
  const int col = (t & 7) * 4, rbase = t >> 3;  // 16 rows/pass, 8 passes
#pragma unroll
  for (int p = 0; p < 8; ++p) {
    const int row = rbase + p * 16;
    const float4 f = *(const float4*)(A + (size_t)(m0 + row) * GK + k0 + col);
    bf16* d = As + row * 40 + col;
    d[0] = (bf16)f.x; d[1] = (bf16)f.y; d[2] = (bf16)f.z; d[3] = (bf16)f.w;
  }
}
__device__ __forceinline__ void stage_a(bf16* As, const bf16* A,
                                        int m0, int k0, int t) {
  const int col = (t & 3) * 8, rbase = t >> 2;  // 32 rows/pass, 4 passes
#pragma unroll
  for (int p = 0; p < 4; ++p) {
    const int row = rbase + p * 32;
    const bf16x8 v = *(const bf16x8*)(A + (size_t)(m0 + row) * GK + k0 + col);
    *(bf16x8*)(As + row * 40 + col) = v;
  }
}
__device__ __forceinline__ void st_c(float* C, size_t i, float v) { C[i] = v; }
__device__ __forceinline__ void st_c(bf16* C, size_t i, float v) { C[i] = (bf16)v; }

template <typename TA, typename TC>
__global__ __launch_bounds__(128) void gemm_bias_kernel(
    const TA* __restrict__ A, const float* __restrict__ W,
    const float* __restrict__ bias, TC* __restrict__ C) {
  constexpr int TM = 128, TN = 128, TK = 32, PA = 40;  // pitch 40 bf16 = 80B
  __shared__ __attribute__((aligned(16))) bf16 As[TM * PA];  // [m][k]
  __shared__ __attribute__((aligned(16))) bf16 Bs[TN * PA];  // [n][k] col-major

  const int m0 = blockIdx.x * TM;
  const int n0 = blockIdx.y * TN;
  const int t = threadIdx.x;
  const int lane = t & 31, wid = t >> 5;
  const int l16 = lane & 15, half = lane >> 4;
  const int wm = wid >> 1, wn = wid & 1;  // 2x2 waves, 64x64 each

  f32x8 acc[4][4];
#pragma unroll
  for (int mt = 0; mt < 4; ++mt)
#pragma unroll
    for (int nt = 0; nt < 4; ++nt) acc[mt][nt] = zero8();

  for (int k0 = 0; k0 < GK; k0 += TK) {
    __syncthreads();
    stage_a(As, A, m0, k0, t);
    {  // stage W tile transposed: Bs[n][k] = W[k0+k][n0+n]
      const int n = (t & 31) * 4, kb = t >> 5;
#pragma unroll
      for (int p = 0; p < 8; ++p) {
        const int k = kb + p * 4;
        const float4 f = *(const float4*)(W + (size_t)(k0 + k) * GN + n0 + n);
        Bs[(n + 0) * PA + k] = (bf16)f.x;
        Bs[(n + 1) * PA + k] = (bf16)f.y;
        Bs[(n + 2) * PA + k] = (bf16)f.z;
        Bs[(n + 3) * PA + k] = (bf16)f.w;
      }
    }
    if (k0 + TK < GK)  // global_prefetch_b8 of next A k-tile
      __builtin_prefetch((const char*)(A + (size_t)(m0 + t) * GK + k0 + TK), 0, 1);
    __syncthreads();

    bf16x16 afr[4], bfr[4];
#pragma unroll
    for (int mt = 0; mt < 4; ++mt) {
      const bf16* base = As + (wm * 64 + mt * 16 + l16) * PA;
      afr[mt] = ld_frag2(base + 16 * half, base + 16 * half + 8);
    }
#pragma unroll
    for (int nt = 0; nt < 4; ++nt) {
      const bf16* base = Bs + (wn * 64 + nt * 16 + l16) * PA + 16 * half;
      bfr[nt] = ld_frag2(base, base + 8);
    }
#pragma unroll
    for (int mt = 0; mt < 4; ++mt)
#pragma unroll
      for (int nt = 0; nt < 4; ++nt)
        acc[mt][nt] = wmma_bf16(afr[mt], bfr[nt], acc[mt][nt]);
  }

#pragma unroll
  for (int mt = 0; mt < 4; ++mt)
#pragma unroll
    for (int nt = 0; nt < 4; ++nt) {
      const int col = n0 + wn * 64 + nt * 16 + l16;
      const float bv = bias[col];
      const int rowb = m0 + wm * 64 + mt * 16 + half * 8;
#pragma unroll
      for (int r = 0; r < 8; ++r)
        st_c(C, (size_t)(rowb + r) * GN + col, acc[mt][nt][r] + bv);
    }
}

// ---------------------------------------------------------------------------
// Flash attention per head (head = contiguous 2048x64 bf16 slice).
// Grid: x = q-tile of 128 rows (16), y = b*H + h (16). Block = 256 thr.
// K/Kf tiles staged by the Tensor Data Mover; V transposed manually.
// ---------------------------------------------------------------------------
__global__ __launch_bounds__(256) void attn_kernel(
    const bf16* __restrict__ q, const bf16* __restrict__ k,
    const bf16* __restrict__ qf, const bf16* __restrict__ kf,
    const bf16* __restrict__ v, bf16* __restrict__ o) {
  constexpr int KT = 64, PH = 72;  // pitch 72 bf16 = 144B
  __shared__ __attribute__((aligned(16))) bf16 Ks[KT * PH];      // [key][d]
  __shared__ __attribute__((aligned(16))) bf16 Kfs[KT * PH];     // [key][d]
  __shared__ __attribute__((aligned(16))) bf16 Vts[DKH * PH];    // [d][key]
  __shared__ __attribute__((aligned(16))) bf16 Ps[8 * 16 * PH];  // per-wave P

  const size_t hb = (size_t)blockIdx.y * (SEQ * DKH);
  const bf16* Q = q + hb;
  const bf16* K = k + hb;
  const bf16* Qf = qf + hb;
  const bf16* Kf = kf + hb;
  const bf16* V = v + hb;
  bf16* O = o + hb;

  const int t = threadIdx.x;
  const int lane = t & 31, wid = t >> 5;
  const int l16 = lane & 15, half = lane >> 4;
  const int qrow = blockIdx.x * 128 + wid * 16 + l16;

  bf16x16 qa[2], qfa[2];
#pragma unroll
  for (int dk = 0; dk < 2; ++dk) {
    const bf16* p = Q + (size_t)qrow * DKH + dk * 32;
    qa[dk] = ld_frag2(p + 16 * half, p + 16 * half + 8);
    const bf16* pf = Qf + (size_t)qrow * DKH + dk * 32;
    qfa[dk] = ld_frag2(pf + 16 * half, pf + 16 * half + 8);
  }

  f32x8 oacc[4];
#pragma unroll
  for (int dt = 0; dt < 4; ++dt) oacc[dt] = zero8();
  float mi[8], li[8];
#pragma unroll
  for (int r = 0; r < 8; ++r) { mi[r] = -3.0e38f; li[r] = 0.0f; }
  const float scale = 0.08838834764831845f;  // 1/sqrt(2*DK)

  for (int kb = 0; kb < SEQ; kb += KT) {
    __syncthreads();
    if (wid == 0) {  // async TDM loads of K / Kf tiles (one wave issues)
      tdm_load_tile64((unsigned)(uintptr_t)Ks, K + (size_t)kb * DKH);
      tdm_load_tile64((unsigned)(uintptr_t)Kfs, Kf + (size_t)kb * DKH);
    }
    // V tile transposed: Vts[d][m] = V[kb+m][d]
#pragma unroll
    for (int p = 0; p < 2; ++p) {
      const int id = t + p * 256;          // 0..511
      const int m = id >> 3;               // key row 0..63
      const int c = (id & 7) * 8;          // d 0..56 step 8
      const bf16x8 fv = *(const bf16x8*)(V + (size_t)(kb + m) * DKH + c);
#pragma unroll
      for (int j = 0; j < 8; ++j) Vts[(c + j) * PH + m] = fv[j];
    }
    if (wid == 0) __builtin_amdgcn_s_wait_tensorcnt(0);
    __syncthreads();

    // S = (Q K^T + Qf Kf^T) for this wave's 16 rows x 64 keys
    f32x8 s[4];
#pragma unroll
    for (int nt = 0; nt < 4; ++nt) {
      s[nt] = zero8();
      const bf16* kr = Ks + (nt * 16 + l16) * PH;
      const bf16* fr = Kfs + (nt * 16 + l16) * PH;
      s[nt] = wmma_bf16(qa[0], ld_frag2(kr + 16 * half, kr + 16 * half + 8), s[nt]);
      s[nt] = wmma_bf16(qa[1], ld_frag2(kr + 32 + 16 * half, kr + 40 + 16 * half), s[nt]);
      s[nt] = wmma_bf16(qfa[0], ld_frag2(fr + 16 * half, fr + 16 * half + 8), s[nt]);
      s[nt] = wmma_bf16(qfa[1], ld_frag2(fr + 32 + 16 * half, fr + 40 + 16 * half), s[nt]);
    }

    // online softmax; rows half*8+r live entirely in this 16-lane half
    float rmax[8], rsum[8], alpha[8];
#pragma unroll
    for (int r = 0; r < 8; ++r) {
      float x = -3.0e38f;
#pragma unroll
      for (int nt = 0; nt < 4; ++nt) {
        s[nt][r] *= scale;
        x = fmaxf(x, s[nt][r]);
      }
      rmax[r] = x;
    }
#pragma unroll
    for (int m = 1; m <= 8; m <<= 1)
#pragma unroll
      for (int r = 0; r < 8; ++r)
        rmax[r] = fmaxf(rmax[r], __shfl_xor(rmax[r], m, 32));
#pragma unroll
    for (int r = 0; r < 8; ++r) {
      const float mn = fmaxf(mi[r], rmax[r]);
      alpha[r] = __expf(mi[r] - mn);
      mi[r] = mn;
      rsum[r] = 0.0f;
    }
#pragma unroll
    for (int nt = 0; nt < 4; ++nt)
#pragma unroll
      for (int r = 0; r < 8; ++r) {
        const float pv = __expf(s[nt][r] - mi[r]);
        s[nt][r] = pv;
        rsum[r] += pv;
      }
#pragma unroll
    for (int m = 1; m <= 8; m <<= 1)
#pragma unroll
      for (int r = 0; r < 8; ++r) rsum[r] += __shfl_xor(rsum[r], m, 32);
#pragma unroll
    for (int r = 0; r < 8; ++r) li[r] = li[r] * alpha[r] + rsum[r];
#pragma unroll
    for (int dt = 0; dt < 4; ++dt)
#pragma unroll
      for (int r = 0; r < 8; ++r) oacc[dt][r] *= alpha[r];

    // spill P (bf16) to this wave's LDS slab (same-wave DS ordering)
    bf16* myP = Ps + wid * 16 * PH;
#pragma unroll
    for (int nt = 0; nt < 4; ++nt)
#pragma unroll
      for (int r = 0; r < 8; ++r)
        myP[(half * 8 + r) * PH + nt * 16 + l16] = (bf16)s[nt][r];

    // O += P @ V
#pragma unroll
    for (int kk = 0; kk < 2; ++kk) {
      const bf16* pr = myP + l16 * PH + kk * 32;
      const bf16x16 pa = ld_frag2(pr + 16 * half, pr + 16 * half + 8);
#pragma unroll
      for (int dt = 0; dt < 4; ++dt) {
        const bf16* vb = Vts + (dt * 16 + l16) * PH + kk * 32 + 16 * half;
        oacc[dt] = wmma_bf16(pa, ld_frag2(vb, vb + 8), oacc[dt]);
      }
    }
  }

  float inv[8];
#pragma unroll
  for (int r = 0; r < 8; ++r) inv[r] = 1.0f / li[r];
  const int rowb = blockIdx.x * 128 + wid * 16 + half * 8;
#pragma unroll
  for (int dt = 0; dt < 4; ++dt)
#pragma unroll
    for (int r = 0; r < 8; ++r)
      O[(size_t)(rowb + r) * DKH + dt * 16 + l16] = (bf16)(oacc[dt][r] * inv[r]);
}

// ---------------------------------------------------------------------------
// Launch
// ---------------------------------------------------------------------------
extern "C" void kernel_launch(void* const* d_in, const int* in_sizes, int n_in,
                              void* d_out, int out_size, void* d_ws, size_t ws_size,
                              hipStream_t stream) {
  (void)in_sizes; (void)n_in; (void)out_size; (void)ws_size;
  const float* x   = (const float*)d_in[0];
  const float* xk  = (const float*)d_in[1];
  const float* Wv  = (const float*)d_in[2];
  const float* bv  = (const float*)d_in[3];
  const float* Wk  = (const float*)d_in[4];
  const float* bk  = (const float*)d_in[5];
  const float* Wq  = (const float*)d_in[6];
  const float* bq  = (const float*)d_in[7];
  const float* Wkf = (const float*)d_in[8];
  const float* bkf = (const float*)d_in[9];
  const float* Wqf = (const float*)d_in[10];
  const float* bqf = (const float*)d_in[11];
  const float* Wq2 = (const float*)d_in[12];
  const float* bq2 = (const float*)d_in[13];
  const float* Wk2 = (const float*)d_in[14];
  const float* bk2 = (const float*)d_in[15];
  const float* Wqf2 = (const float*)d_in[16];
  const float* bqf2 = (const float*)d_in[17];
  const float* Wkf2 = (const float*)d_in[18];
  const float* bkf2 = (const float*)d_in[19];
  const float* Wo  = (const float*)d_in[20];
  const float* bo  = (const float*)d_in[21];
  float* out = (float*)d_out;

  const size_t SZ = (size_t)GM * GN;  // elements per activation buffer
  bf16* ws   = (bf16*)d_ws;
  bf16* vb_  = ws + 0 * SZ;
  bf16* k1   = ws + 1 * SZ;
  bf16* q1   = ws + 2 * SZ;
  bf16* kf1  = ws + 3 * SZ;
  bf16* qf1  = ws + 4 * SZ;
  bf16* q2   = ws + 5 * SZ;
  bf16* k2   = ws + 6 * SZ;
  bf16* qf2  = ws + 7 * SZ;
  bf16* kf2  = ws + 8 * SZ;
  bf16* attn = k1;  // k1 dead after the second projections

  const dim3 gg(GM / 128, GN / 128), gb(128);
  // first projections (f32 activations -> bf16)
  gemm_bias_kernel<float, bf16><<<gg, gb, 0, stream>>>(x,  Wv,  bv,  vb_);
  gemm_bias_kernel<float, bf16><<<gg, gb, 0, stream>>>(x,  Wk,  bk,  k1);
  gemm_bias_kernel<float, bf16><<<gg, gb, 0, stream>>>(x,  Wq,  bq,  q1);
  gemm_bias_kernel<float, bf16><<<gg, gb, 0, stream>>>(xk, Wkf, bkf, kf1);
  gemm_bias_kernel<float, bf16><<<gg, gb, 0, stream>>>(xk, Wqf, bqf, qf1);
  // second projections (bf16 -> bf16)
  gemm_bias_kernel<bf16, bf16><<<gg, gb, 0, stream>>>(q1,  Wq2,  bq2,  q2);
  gemm_bias_kernel<bf16, bf16><<<gg, gb, 0, stream>>>(k1,  Wk2,  bk2,  k2);
  gemm_bias_kernel<bf16, bf16><<<gg, gb, 0, stream>>>(qf1, Wqf2, bqf2, qf2);
  gemm_bias_kernel<bf16, bf16><<<gg, gb, 0, stream>>>(kf1, Wkf2, bkf2, kf2);
  // attention
  attn_kernel<<<dim3(SEQ / 128, 16), 256, 0, stream>>>(q2, k2, qf2, kf2, vb_, attn);
  // output projection (bf16 -> f32)
  gemm_bias_kernel<bf16, float><<<gg, gb, 0, stream>>>(attn, Wo, bo, out);
}
